// TransformerBlock_73143293051032
// MI455X (gfx1250) — matrix operand order repaired and checked by reading, not compiled, not run on hardware
//
#include <hip/hip_runtime.h>
#include <hip/hip_bf16.h>
#include <math.h>

// ---------------------------------------------------------------------------
// Transformer block for MI455X (gfx1250, wave32, WMMA).
// GEMMs / attention matmuls on v_wmma_f32_16x16x32_bf16.
// Tile staging via GLOBAL_LOAD_ASYNC_TO_LDS (ASYNCcnt) and the Tensor Data
// Mover (TENSORcnt); TDM is pipelined one chunk ahead in attention.
// ---------------------------------------------------------------------------

#define EMBED       1024
#define HEADS       16
#define HEAD_DIM    64
#define MLP_HIDDEN  4096
#define SEQ         1024
#define BATCH       8
#define MROWS       (BATCH * SEQ)          // 8192 token rows

#define AS1 __attribute__((address_space(1)))
#define AS3 __attribute__((address_space(3)))

typedef __attribute__((ext_vector_type(16))) __bf16       v16bf;
typedef __attribute__((ext_vector_type(8)))  float        v8f;
typedef __attribute__((ext_vector_type(4)))  unsigned int v4u;
typedef __attribute__((ext_vector_type(4)))  unsigned int v4ui;
typedef __attribute__((ext_vector_type(8)))  int          v8ii;
typedef __attribute__((ext_vector_type(4)))  int          v4ii;
typedef __attribute__((ext_vector_type(4)))  int          v4i;

#if __has_builtin(__builtin_amdgcn_global_load_async_to_lds_b128)
#define HAVE_ASYNC_LDS 1
#else
#define HAVE_ASYNC_LDS 0
#endif

#if __has_builtin(__builtin_amdgcn_tensor_load_to_lds)
#define HAVE_TDM 1
#else
#define HAVE_TDM 0
#endif

union Frag {              // 32 bytes: 16 bf16 elements == 8 VGPRs
  v4u   q[2];
  v16bf v;
};

__device__ __forceinline__ unsigned short f2bf(float f) {
  unsigned int u = __float_as_uint(f);
  u = u + 0x7FFFu + ((u >> 16) & 1u);      // RNE to bf16
  return (unsigned short)(u >> 16);
}

__device__ __forceinline__ v8f v8f_zero() {
  v8f z;
#pragma unroll
  for (int i = 0; i < 8; ++i) z[i] = 0.0f;
  return z;
}

__device__ __forceinline__ v8f wmma_bf16(const Frag& a, const Frag& b, v8f c) {
  return __builtin_amdgcn_wmma_f32_16x16x32_bf16(false, a.v, false, b.v,
                                                 (short)0, c, false, false);
}

__device__ __forceinline__ void async_copy_b128(const unsigned short* g,
                                                unsigned short* l) {
#if HAVE_ASYNC_LDS
  unsigned short* gm = const_cast<unsigned short*>(g);
  __builtin_amdgcn_global_load_async_to_lds_b128(
      (AS1 v4i*)gm, (AS3 v4i*)l, 0, 0);
#endif
}

__device__ __forceinline__ void wait_async_all() {
#if HAVE_ASYNC_LDS
#if __has_builtin(__builtin_amdgcn_s_wait_asynccnt)
  __builtin_amdgcn_s_wait_asynccnt(0);
#else
  asm volatile("s_wait_asynccnt 0" ::: "memory");
#endif
#endif
}

__device__ __forceinline__ void wait_tensor_le0() {
#if __has_builtin(__builtin_amdgcn_s_wait_tensorcnt)
  __builtin_amdgcn_s_wait_tensorcnt(0);
#else
  asm volatile("s_wait_tensorcnt 0" ::: "memory");
#endif
}

__device__ __forceinline__ void wait_tensor_le1() {
#if __has_builtin(__builtin_amdgcn_s_wait_tensorcnt)
  __builtin_amdgcn_s_wait_tensorcnt(1);
#else
  asm volatile("s_wait_tensorcnt 1" ::: "memory");
#endif
}

#if HAVE_TDM
// Issue one 2D TDM descriptor: 64(key) x 64(d) bf16 tile of K, row stride
// 3072 elements, into LDS at lds_off.
__device__ __forceinline__ void attn_issue_k_tdm(const unsigned short* qkv,
                                                 size_t baseTok, int kbase,
                                                 int h, unsigned lds_off) {
  const unsigned long long gaddr = (unsigned long long)(size_t)(
      qkv + (baseTok + kbase) * (size_t)(3 * EMBED) + EMBED + h * HEAD_DIM);
  v4ui g0;
  g0[0] = 1u;                                        // count=1, user D#
  g0[1] = lds_off;                                   // lds_addr (bytes)
  g0[2] = (unsigned)(gaddr & 0xFFFFFFFFull);         // global_addr lo
  g0[3] = (unsigned)((gaddr >> 32) & 0x1FFFFFFull)   // global_addr hi
        | (2u << 30);                                // type=2 (image)
  v8ii g1;
  g1[0] = (int)(1u << 16);        // wg_mask=0, data_size=1 (2 bytes)
  g1[1] = (int)(64u << 16);       // tensor_dim0[15:0]=64 at bits[31:16]
  g1[2] = (int)(64u << 16);       // tensor_dim1[15:0]=64 at bits[31:16]
  g1[3] = (int)(64u << 16);       // tile_dim0=64 at bits[31:16]
  g1[4] = (int)64u;               // tile_dim1=64, tile_dim2=0
  g1[5] = (int)3072;              // tensor_dim0_stride lo32 (elems)
  g1[6] = 0;                      // stride hi, dim1_stride lo
  g1[7] = 0;                      // dim1_stride hi
  v4ii gz4;
  gz4[0] = 0; gz4[1] = 0; gz4[2] = 0; gz4[3] = 0;
  v8ii gz8;
#pragma unroll
  for (int j = 0; j < 8; ++j) gz8[j] = 0;
  __builtin_amdgcn_tensor_load_to_lds(g0, g1, gz4, gz4, gz8, 0);
}
#endif

// ---------------------------------------------------------------------------
// fp32 -> bf16 weight conversion
// ---------------------------------------------------------------------------
__global__ void cvt_bf16_kernel(const float* __restrict__ in,
                                unsigned short* __restrict__ out, int n) {
  int i = blockIdx.x * blockDim.x + threadIdx.x;
  if (i < n) out[i] = f2bf(in[i]);
}

// ---------------------------------------------------------------------------
// LayerNorm over C=1024, emit bf16.  One 256-thread block per row.
// ---------------------------------------------------------------------------
__global__ __launch_bounds__(256)
void ln_bf16_kernel(const float* __restrict__ x,
                    const float* __restrict__ g,
                    const float* __restrict__ bta,
                    unsigned short* __restrict__ out) {
  const int row = blockIdx.x;
  const int t   = threadIdx.x;
  const float* xr = x + (size_t)row * EMBED;

  float4 v = *(const float4*)(xr + t * 4);
  float s = v.x + v.y + v.z + v.w;
  float q = v.x * v.x + v.y * v.y + v.z * v.z + v.w * v.w;

  __shared__ float rs[256];
  __shared__ float rq[256];
  rs[t] = s; rq[t] = q;
  __syncthreads();
  for (int off = 128; off > 0; off >>= 1) {
    if (t < off) { rs[t] += rs[t + off]; rq[t] += rq[t + off]; }
    __syncthreads();
  }
  const float mean = rs[0] * (1.0f / EMBED);
  const float var  = rq[0] * (1.0f / EMBED) - mean * mean;
  const float rinv = rsqrtf(var + 1e-5f);

  float xv[4] = {v.x, v.y, v.z, v.w};
#pragma unroll
  for (int j = 0; j < 4; ++j) {
    int c = t * 4 + j;
    float o = (xv[j] - mean) * rinv * g[c] + bta[c];
    out[(size_t)row * EMBED + c] = f2bf(o);
  }
}

// ---------------------------------------------------------------------------
// bf16 WMMA GEMM: out[M,N] = A[M,K] * B[K,N] + bias (+ epilogue)
// Block tile 128x128, BK=64, double-buffered LDS (64 KB).
// 8 waves (4M x 2N); wave tile 32x64; 16 wmma per stage.
// A staged via GLOBAL_LOAD_ASYNC_TO_LDS; B staged transposed [n][k] through
// VGPRs so B fragments are two contiguous b128 DS reads.
// ---------------------------------------------------------------------------
enum { EP_BF16 = 0, EP_GELU_BF16 = 1, EP_RES_F32 = 2 };

template <int EPI>
__global__ __launch_bounds__(256, 2)
void gemm_bf16_wmma(const unsigned short* __restrict__ A,
                    const unsigned short* __restrict__ Bw,
                    const float* __restrict__ bias,
                    const float* __restrict__ res,
                    void* __restrict__ outp,
                    int M, int N, int K) {
  __shared__ unsigned short sA [2][128 * 64];   // [buf][row][k]
  __shared__ unsigned short sBt[2][128 * 64];   // [buf][n][k]

  const int tid  = threadIdx.x;
  const int lane = tid & 31;
  const int wave = tid >> 5;
  const int wm   = wave & 3;                  // 0..3 -> 32-row slice
  const int wn   = wave >> 2;                 // 0..1 -> 64-col slice
  const int bm   = blockIdx.y * 128;
  const int bn   = blockIdx.x * 128;

  // staging coordinates
  const int ar  = tid >> 1;                   // A row 0..127
  const int ac  = (tid & 1) * 32;             // A k half (32 elems = 4x b128)
  const int bk0 = tid >> 4;                   // B k row base 0..15 (x4 rows)
  const int bnc = (tid & 15) * 8;             // B n chunk (8 elems = 1x b128)

  v8f acc[2][4];
#pragma unroll
  for (int mi = 0; mi < 2; ++mi)
#pragma unroll
    for (int ni = 0; ni < 4; ++ni) acc[mi][ni] = v8f_zero();

  // ISA 16-bit A layout: lanes 0-15 hold K 0-7 & 16-23; lanes 16-31 K 8-15 & 24-31.
  const int arow0 = wm * 32 + (lane & 15);
  const int akb   = (lane >= 16) ? 8 : 0;
  // ISA 16-bit B layout: lanes 0-15 hold K 0-7 & 16-23; lanes 16-31 K 8-15 & 24-31.
  const int bkb   = (lane >= 16) ? 8 : 0;

  const int nT = K >> 6;                      // 64-k tiles

  // ---------------- prologue: stage tile 0 into buffer 0 ----------------
  {
    const size_t abase = (size_t)(bm + ar) * K + ac;
#if HAVE_ASYNC_LDS
#pragma unroll
    for (int j = 0; j < 4; ++j)
      async_copy_b128(A + abase + j * 8, &sA[0][ar * 64 + ac + j * 8]);
#else
    v4u arg[4];
#pragma unroll
    for (int j = 0; j < 4; ++j) arg[j] = *(const v4u*)(A + abase + j * 8);
#pragma unroll
    for (int j = 0; j < 4; ++j) *(v4u*)(&sA[0][ar * 64 + ac + j * 8]) = arg[j];
#endif
    v4u br[4];
#pragma unroll
    for (int h = 0; h < 4; ++h)
      br[h] = *(const v4u*)(Bw + (size_t)(bk0 + h * 16) * N + bn + bnc);
#pragma unroll
    for (int h = 0; h < 4; ++h) {
      const unsigned short* bs = (const unsigned short*)&br[h];
#pragma unroll
      for (int j = 0; j < 8; ++j)
        sBt[0][(bnc + j) * 64 + bk0 + h * 16] = bs[j];
    }
    wait_async_all();
    __syncthreads();
  }

  // ---------------- main loop: compute cur, prefetch nxt ----------------
  for (int t = 0; t < nT; ++t) {
    const int cur = t & 1;
    const int nxt = cur ^ 1;
    const int k0n = (t + 1) << 6;
    const bool more = (t + 1 < nT);

    v4u br[4];
#if !HAVE_ASYNC_LDS
    v4u arg[4];
#endif
    if (more) {
      const size_t abase = (size_t)(bm + ar) * K + k0n + ac;
#if HAVE_ASYNC_LDS
#pragma unroll
      for (int j = 0; j < 4; ++j)
        async_copy_b128(A + abase + j * 8, &sA[nxt][ar * 64 + ac + j * 8]);
#else
#pragma unroll
      for (int j = 0; j < 4; ++j) arg[j] = *(const v4u*)(A + abase + j * 8);
#endif
#pragma unroll
      for (int h = 0; h < 4; ++h)
        br[h] = *(const v4u*)(Bw + (size_t)(k0n + bk0 + h * 16) * N + bn + bnc);
      if (t + 2 < nT)   // L2 warm-up for the tile after next
        __builtin_prefetch(Bw + (size_t)(k0n + 64 + bk0) * N + bn + bnc, 0, 0);
    }

    // ---- 16 WMMAs on the current buffer (two 32-k steps) ----
#pragma unroll
    for (int s = 0; s < 2; ++s) {
      Frag af[2];
#pragma unroll
      for (int mi = 0; mi < 2; ++mi) {
        int row = arow0 + mi * 16;
        af[mi].q[0] = *(const v4u*)(&sA[cur][row * 64 + s * 32 + akb]);
        af[mi].q[1] = *(const v4u*)(&sA[cur][row * 64 + s * 32 + akb + 16]);
      }
#pragma unroll
      for (int ni = 0; ni < 4; ++ni) {
        int col = wn * 64 + ni * 16 + (lane & 15);
        Frag bfr;
        bfr.q[0] = *(const v4u*)(&sBt[cur][col * 64 + s * 32 + bkb]);
        bfr.q[1] = *(const v4u*)(&sBt[cur][col * 64 + s * 32 + bkb + 16]);
#pragma unroll
        for (int mi = 0; mi < 2; ++mi)
          acc[mi][ni] = wmma_bf16(af[mi], bfr, acc[mi][ni]);
      }
    }

    if (more) {
#pragma unroll
      for (int h = 0; h < 4; ++h) {
        const unsigned short* bs = (const unsigned short*)&br[h];
#pragma unroll
        for (int j = 0; j < 8; ++j)
          sBt[nxt][(bnc + j) * 64 + bk0 + h * 16] = bs[j];
      }
#if !HAVE_ASYNC_LDS
#pragma unroll
      for (int j = 0; j < 4; ++j)
        *(v4u*)(&sA[nxt][ar * 64 + ac + j * 8]) = arg[j];
#endif
    }
    wait_async_all();
    __syncthreads();
  }

  // ---------------- epilogue ----------------
  unsigned short* ob = (unsigned short*)outp;
  float*          of = (float*)outp;
  const int rowh = (lane >= 16) ? 8 : 0;
#pragma unroll
  for (int mi = 0; mi < 2; ++mi) {
    int rbase = bm + wm * 32 + mi * 16 + rowh;
#pragma unroll
    for (int ni = 0; ni < 4; ++ni) {
      int col = bn + wn * 64 + ni * 16 + (lane & 15);
      float bc = bias[col];
#pragma unroll
      for (int i = 0; i < 8; ++i) {
        size_t o = (size_t)(rbase + i) * N + col;
        float v = acc[mi][ni][i] + bc;
        if (EPI == EP_GELU_BF16) v = 0.5f * v * (1.0f + erff(v * 0.70710678f));
        if (EPI == EP_RES_F32) of[o] = v + res[o];
        else                   ob[o] = f2bf(v);
      }
    }
  }
}

// ---------------------------------------------------------------------------
// Flash-style attention.  grid = (SEQ/64, HEADS, BATCH), 128 threads (4 waves).
// Each wave owns 16 query rows; keys in chunks of 64 with online softmax.
// K tiles are staged by the Tensor Data Mover into a double buffer, pipelined
// one chunk ahead (s_wait_tensorcnt 1 exploits in-order TDM completion).
// V staged transposed by the block.  P converts to bf16 via a per-wave LDS
// round trip (C layout -> A-fragment layout).
// ---------------------------------------------------------------------------
__global__ __launch_bounds__(128, 2)
void attn_kernel(const unsigned short* __restrict__ qkv,   // [B,N,3,H,D] bf16
                 unsigned short* __restrict__ out) {       // [B,N,H*D]   bf16
  const int tid  = threadIdx.x;
  const int lane = tid & 31;
  const int wave = tid >> 5;                 // 0..3
  const int qblk = blockIdx.x;               // 64-query block
  const int h    = blockIdx.y;
  const int b    = blockIdx.z;

  __shared__ unsigned short sK [2][64 * 64]; // [buf][key][d]
  __shared__ unsigned short sVt[64 * 64];    // [d][key]
  __shared__ unsigned short sP [4][16 * 64]; // per-wave P tile [row][key]

  const size_t tokStride = 3 * EMBED;        // 3072 bf16 per token
  const size_t baseTok   = (size_t)b * SEQ;

  // Q fragments (constant across the key loop).
  const int q0 = qblk * 64 + wave * 16;
  Frag qf[2];
  {
    int row = q0 + (lane & 15);
    int kb  = (lane >= 16) ? 8 : 0;
    const unsigned short* qp = qkv + (baseTok + row) * tokStride + h * HEAD_DIM;
#pragma unroll
    for (int s = 0; s < 2; ++s) {
      qf[s].q[0] = *(const v4u*)(qp + s * 32 + kb);
      qf[s].q[1] = *(const v4u*)(qp + s * 32 + kb + 16);
    }
  }

  float mrow[8], lrow[8];
#pragma unroll
  for (int i = 0; i < 8; ++i) { mrow[i] = -1e30f; lrow[i] = 0.0f; }
  v8f oacc[4];
#pragma unroll
  for (int nt = 0; nt < 4; ++nt) oacc[nt] = v8f_zero();

  const int nC = SEQ / 64;

#if HAVE_TDM
  // Prologue: kick off TDM for chunk 0 into buffer 0.
  if (wave == 0)
    attn_issue_k_tdm(qkv, baseTok, 0, h,
                     (unsigned)(size_t)(AS3 unsigned short*)&sK[0][0]);
#endif

  for (int c = 0; c < nC; ++c) {
    const int kbase = c * 64;
    const int cur = c & 1;
    const int nxt = cur ^ 1;
    const bool more = (c + 1 < nC);
    __syncthreads();                         // previous chunk done with LDS

#if HAVE_TDM
    if (wave == 0 && more)                   // pipeline: chunk c+1 into nxt
      attn_issue_k_tdm(qkv, baseTok, kbase + 64, h,
                       (unsigned)(size_t)(AS3 unsigned short*)&sK[nxt][0]);
#endif
    { // ---- stage V transposed [d][key]; 2 threads per key row ----
      int r  = tid >> 1;                     // key 0..63
      int c0 = (tid & 1) * 32;               // d half
      const unsigned short* vp =
          qkv + (baseTok + kbase + r) * tokStride + 2 * EMBED + h * HEAD_DIM + c0;
      v4u vv[4];
#pragma unroll
      for (int t4 = 0; t4 < 4; ++t4) vv[t4] = *(const v4u*)(vp + t4 * 8);
      const unsigned short* vs = (const unsigned short*)vv;
#pragma unroll
      for (int j = 0; j < 32; ++j) sVt[(c0 + j) * 64 + r] = vs[j];
#if !HAVE_TDM
      const unsigned short* kp =
          qkv + (baseTok + kbase + r) * tokStride + EMBED + h * HEAD_DIM + c0;
      *(v4u*)(&sK[cur][r * 64 + c0])      = *(const v4u*)(kp);
      *(v4u*)(&sK[cur][r * 64 + c0 + 8])  = *(const v4u*)(kp + 8);
      *(v4u*)(&sK[cur][r * 64 + c0 + 16]) = *(const v4u*)(kp + 16);
      *(v4u*)(&sK[cur][r * 64 + c0 + 24]) = *(const v4u*)(kp + 24);
#endif
    }
#if HAVE_TDM
    if (wave == 0) {
      // In-order TDM completion: <=1 outstanding means chunk c has landed.
      if (more) wait_tensor_le1(); else wait_tensor_le0();
    }
#endif
    __syncthreads();

    // ---- S = scale * Q K^T  (contraction over d=64 -> two k-steps) ----
    v8f sacc[4];
#pragma unroll
    for (int nt = 0; nt < 4; ++nt) sacc[nt] = v8f_zero();
    {
      int key = (lane & 15);
      int bkb = (lane >= 16) ? 8 : 0;
#pragma unroll
      for (int nt = 0; nt < 4; ++nt) {
        int kr = (nt * 16 + key) * 64;
#pragma unroll
        for (int s = 0; s < 2; ++s) {
          Frag kf;
          kf.q[0] = *(const v4u*)(&sK[cur][kr + s * 32 + bkb]);
          kf.q[1] = *(const v4u*)(&sK[cur][kr + s * 32 + bkb + 16]);
          sacc[nt] = wmma_bf16(qf[s], kf, sacc[nt]);
        }
      }
    }
#pragma unroll
    for (int nt = 0; nt < 4; ++nt)
#pragma unroll
      for (int i = 0; i < 8; ++i) sacc[nt][i] *= 0.125f;   // 1/sqrt(64)

    // ---- online softmax: row stats via 16-lane-half shfl reductions ----
    float mc[8], alpha[8], rsum[8];
#pragma unroll
    for (int i = 0; i < 8; ++i) {
      float v = fmaxf(fmaxf(sacc[0][i], sacc[1][i]),
                      fmaxf(sacc[2][i], sacc[3][i]));
#pragma unroll
      for (int d = 1; d < 16; d <<= 1) v = fmaxf(v, __shfl_xor(v, d, 32));
      mc[i] = v;
      float mnew = fmaxf(mrow[i], mc[i]);
      alpha[i] = __expf(mrow[i] - mnew);
      mrow[i] = mnew;
      rsum[i] = 0.0f;
    }
#pragma unroll
    for (int nt = 0; nt < 4; ++nt)
#pragma unroll
      for (int i = 0; i < 8; ++i) {
        float p = __expf(sacc[nt][i] - mrow[i]);
        sacc[nt][i] = p;
        rsum[i] += p;
      }
#pragma unroll
    for (int i = 0; i < 8; ++i) {
#pragma unroll
      for (int d = 1; d < 16; d <<= 1) rsum[i] += __shfl_xor(rsum[i], d, 32);
      lrow[i] = lrow[i] * alpha[i] + rsum[i];
    }
#pragma unroll
    for (int nt = 0; nt < 4; ++nt)
#pragma unroll
      for (int i = 0; i < 8; ++i) oacc[nt][i] *= alpha[i];

    // ---- P (C layout) -> LDS -> A-fragment layout ----
    {
      int rh = (lane >= 16) ? 8 : 0;
      int cc = (lane & 15);
#pragma unroll
      for (int nt = 0; nt < 4; ++nt)
#pragma unroll
        for (int i = 0; i < 8; ++i)
          sP[wave][(i + rh) * 64 + nt * 16 + cc] = f2bf(sacc[nt][i]);
    }
    __syncthreads();   // order P store vs P load (uniform across waves)

    Frag pf[2];
    {
      int prow = (lane & 15);
      int pkb  = (lane >= 16) ? 8 : 0;
#pragma unroll
      for (int s = 0; s < 2; ++s) {
        pf[s].q[0] = *(const v4u*)(&sP[wave][prow * 64 + s * 32 + pkb]);
        pf[s].q[1] = *(const v4u*)(&sP[wave][prow * 64 + s * 32 + pkb + 16]);
      }
    }
    // ---- O += P * V  (contraction over 64 keys -> two k-steps) ----
    {
      int vkb = (lane >= 16) ? 8 : 0;
#pragma unroll
      for (int nt = 0; nt < 4; ++nt) {
        int dr = (nt * 16 + (lane & 15)) * 64;
#pragma unroll
        for (int s = 0; s < 2; ++s) {
          Frag vf;
          vf.q[0] = *(const v4u*)(&sVt[dr + s * 32 + vkb]);
          vf.q[1] = *(const v4u*)(&sVt[dr + s * 32 + vkb + 16]);
          oacc[nt] = wmma_bf16(pf[s], vf, oacc[nt]);
        }
      }
    }
  }

  // ---- normalize and write O as [B,N,H*D] bf16 ----
  const int rh = (lane >= 16) ? 8 : 0;
#pragma unroll
  for (int nt = 0; nt < 4; ++nt) {
    int dcol = nt * 16 + (lane & 15);
#pragma unroll
    for (int i = 0; i < 8; ++i) {
      int qrow = q0 + i + rh;
      float o = oacc[nt][i] / lrow[i];
      out[(baseTok + qrow) * EMBED + h * HEAD_DIM + dcol] = f2bf(o);
    }
  }
}

// ---------------------------------------------------------------------------
// Host-side orchestration
// ---------------------------------------------------------------------------
extern "C" void kernel_launch(void* const* d_in, const int* in_sizes, int n_in,
                              void* d_out, int out_size, void* d_ws, size_t ws_size,
                              hipStream_t stream) {
  const float* x      = (const float*)d_in[0];
  const float* qkv_w  = (const float*)d_in[1];
  const float* qkv_b  = (const float*)d_in[2];
  const float* proj_w = (const float*)d_in[3];
  const float* proj_b = (const float*)d_in[4];
  const float* fc1_w  = (const float*)d_in[5];
  const float* fc1_b  = (const float*)d_in[6];
  const float* fc2_w  = (const float*)d_in[7];
  const float* fc2_b  = (const float*)d_in[8];
  const float* ln1_g  = (const float*)d_in[9];
  const float* ln1_b  = (const float*)d_in[10];
  const float* ln2_g  = (const float*)d_in[11];
  const float* ln2_b  = (const float*)d_in[12];

  char* ws = (char*)d_ws;
  size_t off = 0;
  auto alloc_u16 = [&](size_t n) {
    unsigned short* p = (unsigned short*)(ws + off);
    off += ((n * 2 + 255) / 256) * 256;
    return p;
  };
  auto alloc_f32 = [&](size_t n) {
    float* p = (float*)(ws + off);
    off += ((n * 4 + 255) / 256) * 256;
    return p;
  };

  unsigned short* wqkv  = alloc_u16((size_t)EMBED * 3 * EMBED);
  unsigned short* wproj = alloc_u16((size_t)EMBED * EMBED);
  unsigned short* wfc1  = alloc_u16((size_t)EMBED * MLP_HIDDEN);
  unsigned short* wfc2  = alloc_u16((size_t)MLP_HIDDEN * EMBED);
  unsigned short* h1    = alloc_u16((size_t)MROWS * EMBED);
  unsigned short* qkv   = alloc_u16((size_t)MROWS * 3 * EMBED);
  unsigned short* ao    = alloc_u16((size_t)MROWS * EMBED);
  float*          x1    = alloc_f32((size_t)MROWS * EMBED);
  unsigned short* h2    = alloc_u16((size_t)MROWS * EMBED);
  unsigned short* mact  = alloc_u16((size_t)MROWS * MLP_HIDDEN);

  auto cvt = [&](const float* src, unsigned short* dst, int n) {
    cvt_bf16_kernel<<<(n + 255) / 256, 256, 0, stream>>>(src, dst, n);
  };
  cvt(qkv_w,  wqkv,  EMBED * 3 * EMBED);
  cvt(proj_w, wproj, EMBED * EMBED);
  cvt(fc1_w,  wfc1,  EMBED * MLP_HIDDEN);
  cvt(fc2_w,  wfc2,  MLP_HIDDEN * EMBED);

  // h1 = LN1(x) in bf16
  ln_bf16_kernel<<<MROWS, 256, 0, stream>>>(x, ln1_g, ln1_b, h1);

  // qkv = h1 @ Wqkv + b           [8192 x 3072]
  gemm_bf16_wmma<EP_BF16><<<dim3(3 * EMBED / 128, MROWS / 128), 256, 0, stream>>>(
      h1, wqkv, qkv_b, nullptr, qkv, MROWS, 3 * EMBED, EMBED);

  // attention -> ao (bf16, [B,N,H*D])
  attn_kernel<<<dim3(SEQ / 64, HEADS, BATCH), 128, 0, stream>>>(qkv, ao);

  // x1 = x + ao @ Wproj + b       (fp32 residual stream)
  gemm_bf16_wmma<EP_RES_F32><<<dim3(EMBED / 128, MROWS / 128), 256, 0, stream>>>(
      ao, wproj, proj_b, x, x1, MROWS, EMBED, EMBED);

  // h2 = LN2(x1) in bf16
  ln_bf16_kernel<<<MROWS, 256, 0, stream>>>(x1, ln2_g, ln2_b, h2);

  // mact = gelu(h2 @ Wfc1 + b)    [8192 x 4096] bf16
  gemm_bf16_wmma<EP_GELU_BF16><<<dim3(MLP_HIDDEN / 128, MROWS / 128), 256, 0, stream>>>(
      h2, wfc1, fc1_b, nullptr, mact, MROWS, MLP_HIDDEN, EMBED);

  // out = x1 + mact @ Wfc2 + b    (fp32)
  gemm_bf16_wmma<EP_RES_F32><<<dim3(EMBED / 128, MROWS / 128), 256, 0, stream>>>(
      mact, wfc2, fc2_b, x1, d_out, MROWS, EMBED, MLP_HIDDEN);
}
